// GMMNet_20822001451602
// MI455X (gfx1250) — matrix operand order repaired and checked by reading, not compiled, not run on hardware
//
#include <hip/hip_runtime.h>

// ---------------------------------------------------------------------------
// GMM scan for MI455X (gfx1250, wave32).
// Per-pixel GMM recurrence; the 4 tiny MLPs are executed on the WMMA pipe as
// v_wmma_f32_16x16x32_f16 (M=Cout pad16, K=Cin pad32, N=16 pixels/column-grp),
// with weights pre-staged in A-layout registers and biases as the C operand.
// Per-pixel nonlinearities (density, softmax, sigmoid, exp) stay in fp32 VALU,
// transposed through per-wave LDS slices (80B strides -> conflict-free b128).
// ---------------------------------------------------------------------------

typedef __attribute__((ext_vector_type(16))) _Float16 v16h;
typedef __attribute__((ext_vector_type(8)))  _Float16 v8h;
typedef __attribute__((ext_vector_type(8)))  float    v8f;

#define DEV __device__ __forceinline__

constexpr int BB  = 4, SS = 8, CCh = 3, HH = 384, WWd = 384, KK = 5, CKK = 15;
constexpr int HWX  = HH * WWd;       // 147456
constexpr int NPIX = BB * HWX;       // 589824
constexpr int TPB  = 256;
constexpr int NBLK = NPIX / TPB;     // 2304 (exact)

// per-wave LDS layout: 3 buffers of 32 pixel-rows x 80 bytes
constexpr int IN_STRIDE_H  = 40;     // halfs  (80 B/row)
constexpr int HID_STRIDE_H = 40;     // halfs  (80 B/row; halfs 16..39 stay 0)
constexpr int OUT_STRIDE_F = 20;     // floats (80 B/row)
constexpr int WAVE_LDS  = 32 * 80 * 3;       // 7680 B
constexpr int LDS_BYTES = WAVE_LDS * (TPB/32); // 61440 B

DEV void wait_lds() { asm volatile("s_wait_dscnt 0" ::: "memory"); }

// Stage a (Cout x Cin) fp32 weight matrix as the WMMA A operand (contiguous k order, B uses the same):
// lane: row M = lane&15; element e -> K = e + 16*(lane>=16).
DEV v16h loadA(const float* __restrict__ w, int Cout, int Cin, int lane) {
  const int m  = lane & 15;
  const int hi = lane >> 4;
  v16h a;
#pragma unroll
  for (int e = 0; e < 16; ++e) {
    const int k = e + 16 * hi;
    float v = 0.0f;
    if (m < Cout && k < Cin) v = w[m * Cin + k];
    a[e] = (_Float16)v;
  }
  return a;
}

// Stage a bias vector in C/D-matrix layout: reg r holds bias[r + 8*(lane>=16)].
DEV v8f loadC(const float* __restrict__ b, int Cout, int lane) {
  const int hi = lane >> 4;
  v8f c;
#pragma unroll
  for (int r = 0; r < 8; ++r) {
    const int m = r + 8 * hi;
    c[r] = (m < Cout) ? b[m] : 0.0f;
  }
  return c;
}

// Two-layer MLP on 32 pixels (one wave): in[CIN] per lane -> out[COUT] per lane.
// Layer1: D = relu(W1 * X + b1) via WMMA; Layer2: D = W2 * H + b2 via WMMA.
template <int CIN, int CHID, int COUT>
DEV void run_mlp(const float* in,
                 v16h A1, v8f C1, v16h A2, v8f C2,
                 _Float16* inBuf, _Float16* hidBuf, float* outBuf,
                 int lane, float* out) {
  const int n  = lane & 15;
  const int hi = lane >> 4;

  // stage this lane's input vector, zero-padded to K=32, as f16
  {
    v8h pk[4];
#pragma unroll
    for (int j = 0; j < 32; ++j) {
      const float v = (j < CIN) ? in[j] : 0.0f;
      pk[j >> 3][j & 7] = (_Float16)v;
    }
    v8h* dst = (v8h*)(inBuf + lane * IN_STRIDE_H);
    dst[0] = pk[0]; dst[1] = pk[1]; dst[2] = pk[2]; dst[3] = pk[3];
  }
  wait_lds();

  // ---- layer 1: two 16-pixel column groups ----
  v8f d[2];
#pragma unroll
  for (int g = 0; g < 2; ++g) {
    const int p = g * 16 + n;
    const v8h* src = (const v8h*)(inBuf + p * IN_STRIDE_H + hi * 16);
    union { v16h v; v8h h[2]; } u;
    u.h[0] = src[0]; u.h[1] = src[1];
    d[g] = __builtin_amdgcn_wmma_f32_16x16x32_f16(
        false, A1, false, u.v, (short)0, C1, false, false);
  }
  // relu + scatter hidden (channels 0..15 only; K=16..31 region stays zero)
#pragma unroll
  for (int g = 0; g < 2; ++g) {
    v8h hv;
#pragma unroll
    for (int r = 0; r < 8; ++r) hv[r] = (_Float16)fmaxf(d[g][r], 0.0f);
    *(v8h*)(hidBuf + (g * 16 + n) * HID_STRIDE_H + hi * 8) = hv;
  }
  wait_lds();

  // ---- layer 2 ----
#pragma unroll
  for (int g = 0; g < 2; ++g) {
    const int p = g * 16 + n;
    const v8h* src = (const v8h*)(hidBuf + p * HID_STRIDE_H + hi * 16);
    union { v16h v; v8h h[2]; } u;
    u.h[0] = src[0]; u.h[1] = src[1];   // hi lanes read the permanent-zero pad
    v8f d2 = __builtin_amdgcn_wmma_f32_16x16x32_f16(
        false, A2, false, u.v, (short)0, C2, false, false);
    float* dst = outBuf + p * OUT_STRIDE_F + hi * 8;
#pragma unroll
    for (int r = 0; r < 8; r += 4) *(float4*)(dst + r) =
        make_float4(d2[r], d2[r + 1], d2[r + 2], d2[r + 3]);
  }
  wait_lds();

  // gather this lane's pixel result
#pragma unroll
  for (int j = 0; j < COUT; ++j) out[j] = outBuf[lane * OUT_STRIDE_F + j];
}

// Gaussian density, C=3: (2*pi*s^2)^(-3/2) * exp(-dist / (2 s^2))
DEV void densityK(const float x[3], const float* mu, const float* sg, float* dens) {
#pragma unroll
  for (int k = 0; k < KK; ++k) {
    const float s2 = sg[k] * sg[k];
    const float d0 = x[0] - mu[3 * k + 0];
    const float d1 = x[1] - mu[3 * k + 1];
    const float d2 = x[2] - mu[3 * k + 2];
    const float dist = d0 * d0 + d1 * d1 + d2 * d2;
    const float r = rsqrtf(6.2831853071795864f * s2);  // (2 pi s2)^-1/2
    const float inv2s2 = 3.1415926535897932f * (r * r); // = 1/(2 s2)
    dens[k] = (r * r * r) * __expf(-dist * inv2s2);
  }
}

__global__ __launch_bounds__(TPB) void gmm_scan_kernel(
    const float* __restrict__ frames, const float* __restrict__ mu0,
    const float* __restrict__ pw1, const float* __restrict__ pb1,
    const float* __restrict__ pw2, const float* __restrict__ pb2,
    const float* __restrict__ mw1, const float* __restrict__ mb1,
    const float* __restrict__ mw2, const float* __restrict__ mb2,
    const float* __restrict__ sw1, const float* __restrict__ sb1,
    const float* __restrict__ sw2, const float* __restrict__ sb2,
    const float* __restrict__ gw1, const float* __restrict__ gb1,
    const float* __restrict__ gw2, const float* __restrict__ gb2,
    float* __restrict__ outp) {
  __shared__ __align__(16) unsigned char smem[LDS_BYTES];
  const int tid = threadIdx.x;

  // zero LDS once: the hidden buffer's K=16..31 pad must be 0 forever
  {
    uint4 zz; zz.x = zz.y = zz.z = zz.w = 0u;
    uint4* zp = (uint4*)smem;
#pragma unroll
    for (int i = 0; i < LDS_BYTES / 16 / TPB; ++i) zp[tid + i * TPB] = zz;
  }
  __syncthreads();

  const int lane = tid & 31;
  const int wave = tid >> 5;
  _Float16* inBuf  = (_Float16*)(smem + wave * WAVE_LDS);
  _Float16* hidBuf = (_Float16*)(smem + wave * WAVE_LDS + 2560);
  float*    outBuf = (float*)   (smem + wave * WAVE_LDS + 5120);

  const int q  = blockIdx.x * TPB + tid;   // pixel id, exact coverage
  const int b  = q / HWX;
  const int hw = q - b * HWX;

  // ---- stage weights (A-layout) and biases (C-layout) once per block ----
  const v16h Ap1 = loadA(pw1, KK, 2 * KK, lane),          Ap2 = loadA(pw2, KK, KK, lane);
  const v16h Am1 = loadA(mw1, CKK, CCh + CKK + KK, lane), Am2 = loadA(mw2, CKK, CKK, lane);
  const v16h As1 = loadA(sw1, KK, CCh + CKK + KK, lane),  As2 = loadA(sw2, KK, KK, lane);
  const v16h Ag1 = loadA(gw1, KK, KK, lane),              Ag2 = loadA(gw2, 1, KK, lane);
  const v8f Cp1 = loadC(pb1, KK, lane),  Cp2 = loadC(pb2, KK, lane);
  const v8f Cm1 = loadC(mb1, CKK, lane), Cm2 = loadC(mb2, CKK, lane);
  const v8f Cs1 = loadC(sb1, KK, lane),  Cs2 = loadC(sb2, KK, lane);
  const v8f Cg1 = loadC(gb1, KK, lane),  Cg2 = loadC(gb2, 1, lane);

  // ---- per-pixel state in registers ----
  float pi[KK], sigma[KK], mu[CKK];
#pragma unroll
  for (int k = 0; k < KK; ++k) { pi[k] = 1.0f / KK; sigma[k] = 1.0f; }
#pragma unroll
  for (int j = 0; j < CKK; ++j) mu[j] = mu0[(b * CKK + j) * HWX + hw];

  for (int s = 0; s < SS; ++s) {
    const float* fr = frames + (b * SS + s) * CCh * HWX + hw;
    float x[3] = { fr[0], fr[HWX], fr[2 * HWX] };
    if (s + 1 < SS) {                    // prefetch next frame planes
      const float* nf = fr + CCh * HWX;
      __builtin_prefetch(nf, 0, 1);
      __builtin_prefetch(nf + HWX, 0, 1);
      __builtin_prefetch(nf + 2 * HWX, 0, 1);
    }

    float dens[KK], rho[KK], in[23], z5[KK];
    densityK(x, mu, sigma, dens);
#pragma unroll
    for (int k = 0; k < KK; ++k) {
      const float a = pi[k] * dens[k];   // alpha
      in[k] = pi[k]; in[KK + k] = a;
      rho[k] = a * dens[k];
    }

    // pi_new = softmax(MLP_pi([pi, alpha]))
    run_mlp<2 * KK, KK, KK>(in, Ap1, Cp1, Ap2, Cp2, inBuf, hidBuf, outBuf, lane, z5);
    {
      float mx = z5[0];
#pragma unroll
      for (int k = 1; k < KK; ++k) mx = fmaxf(mx, z5[k]);
      float sum = 0.0f, e[KK];
#pragma unroll
      for (int k = 0; k < KK; ++k) { e[k] = __expf(z5[k] - mx); sum += e[k]; }
      const float inv = 1.0f / sum;
#pragma unroll
      for (int k = 0; k < KK; ++k) pi[k] = e[k] * inv;
    }

    // mu_new = sigmoid(MLP_mu([x, mu, rho]))
    in[0] = x[0]; in[1] = x[1]; in[2] = x[2];
#pragma unroll
    for (int j = 0; j < CKK; ++j) in[3 + j] = mu[j];
#pragma unroll
    for (int k = 0; k < KK; ++k) in[3 + CKK + k] = rho[k];
    float z15[CKK], munew[CKK];
    run_mlp<23, CKK, CKK>(in, Am1, Cm1, Am2, Cm2, inBuf, hidBuf, outBuf, lane, z15);
#pragma unroll
    for (int j = 0; j < CKK; ++j) munew[j] = 1.0f / (1.0f + __expf(-z15[j]));

    // sigma_new = exp(relu(MLP_sigma([x, mu_new, rho])))
#pragma unroll
    for (int j = 0; j < CKK; ++j) in[3 + j] = munew[j];
    run_mlp<23, KK, KK>(in, As1, Cs1, As2, Cs2, inBuf, hidBuf, outBuf, lane, z5);
    float signew[KK];
#pragma unroll
    for (int k = 0; k < KK; ++k) signew[k] = __expf(fmaxf(z5[k], 0.0f));

    // gamma = sigmoid(MLP_gamma(pi_new * dens2))
    densityK(x, munew, signew, dens);
#pragma unroll
    for (int k = 0; k < KK; ++k) in[k] = pi[k] * dens[k];
    float z1[1];
    run_mlp<KK, KK, 1>(in, Ag1, Cg1, Ag2, Cg2, inBuf, hidBuf, outBuf, lane, z1);
    outp[(b * SS + s) * HWX + hw] = 1.0f / (1.0f + __expf(-z1[0]));

#pragma unroll
    for (int j = 0; j < CKK; ++j) mu[j] = munew[j];
#pragma unroll
    for (int k = 0; k < KK; ++k) sigma[k] = signew[k];
  }
}

extern "C" void kernel_launch(void* const* d_in, const int* in_sizes, int n_in,
                              void* d_out, int out_size, void* d_ws, size_t ws_size,
                              hipStream_t stream) {
  (void)in_sizes; (void)n_in; (void)d_ws; (void)ws_size; (void)out_size;
  const float* frames = (const float*)d_in[0];
  // d_in[1] = targets (unused by reference output)
  const float* mu0 = (const float*)d_in[2];
  const float* pw1 = (const float*)d_in[3];
  const float* pb1 = (const float*)d_in[4];
  const float* pw2 = (const float*)d_in[5];
  const float* pb2 = (const float*)d_in[6];
  const float* mw1 = (const float*)d_in[7];
  const float* mb1 = (const float*)d_in[8];
  const float* mw2 = (const float*)d_in[9];
  const float* mb2 = (const float*)d_in[10];
  const float* sw1 = (const float*)d_in[11];
  const float* sb1 = (const float*)d_in[12];
  const float* sw2 = (const float*)d_in[13];
  const float* sb2 = (const float*)d_in[14];
  const float* gw1 = (const float*)d_in[15];
  const float* gb1 = (const float*)d_in[16];
  const float* gw2 = (const float*)d_in[17];
  const float* gb2 = (const float*)d_in[18];

  gmm_scan_kernel<<<NBLK, TPB, 0, stream>>>(
      frames, mu0, pw1, pb1, pw2, pb2, mw1, mb1, mw2, mb2,
      sw1, sb1, sw2, sb2, gw1, gb1, gw2, gb2, (float*)d_out);
}
